// CausalSelfAttention_31473520345485
// MI455X (gfx1250) — compile-verified, matrix operand order checked by reading, not run on hardware
//
#include <hip/hip_runtime.h>
#include <hip/hip_bf16.h>

typedef __attribute__((ext_vector_type(16))) __bf16 v16bf;
typedef __attribute__((ext_vector_type(8)))  __bf16 v8bf;
typedef __attribute__((ext_vector_type(8)))  float  v8f;
typedef __attribute__((ext_vector_type(4)))  float  v4f;

constexpr int Bc  = 4;
constexpr int Tc  = 2048;
constexpr int Cc  = 1024;
constexpr int NHc = 16;
constexpr int HDc = 64;
constexpr int Mrows = Bc * Tc;          // 8192
constexpr int N3 = 3 * Cc;              // 3072

static __device__ inline v16bf combine16(v8bf lo, v8bf hi) {
  return __builtin_shufflevector(lo, hi, 0,1,2,3,4,5,6,7,8,9,10,11,12,13,14,15);
}

static __device__ inline v8bf cvt8(v4f a, v4f b) {
  v8bf r;
  r[0]=(__bf16)a[0]; r[1]=(__bf16)a[1]; r[2]=(__bf16)a[2]; r[3]=(__bf16)a[3];
  r[4]=(__bf16)b[0]; r[5]=(__bf16)b[1]; r[6]=(__bf16)b[2]; r[7]=(__bf16)b[3];
  return r;
}

// ---------------------------------------------------------------------------
// Tiled WMMA GEMM: out = A(fp32, MxK) @ W(fp32, KxN) + bias.
// MODE 0: store fp32 row-major to outF.
// MODE 1: scatter QKV -> Qb/Kb [b,h,t,64] bf16, Vt [b,h,d,T] bf16.
// Workgroup: 256 threads (8 waves), tile 128(M) x 128(N), K-step 32,
// double-buffered global->register staging (loads overlap the 8 WMMAs).
// ---------------------------------------------------------------------------
template<int MODE>
__global__ __launch_bounds__(256)
void wmma_gemm_kernel(const float* __restrict__ A, const float* __restrict__ W,
                      const float* __restrict__ bias, float* __restrict__ outF,
                      __bf16* __restrict__ Qb, __bf16* __restrict__ Kb,
                      __bf16* __restrict__ Vt,
                      int M, int N, int K)
{
  __shared__ alignas(16) __bf16 lA[128][40];    // row-major, 80B rows
  __shared__ alignas(16) __bf16 lBt[128][40];   // transposed: [n][k]

  const int tid  = threadIdx.x;
  const int lane = tid & 31;
  const int wave = tid >> 5;
  const int n0 = blockIdx.x * 128;
  const int m0 = blockIdx.y * 128;
  const int col  = lane & 15;
  const int hsel = (lane >> 4) << 3;            // 0 / 8
  const int mrow = 16 * wave + col;

  // staging assignments
  const int rowA = tid >> 1;                    // 0..127
  const int c0   = (tid & 1) * 16;              // 0 / 16
  const int kr   = tid >> 3;                    // 0..31
  const int nb   = (tid & 7) * 16;              // 0..112

  const float* pa = A + (size_t)(m0 + rowA) * K + c0;
  const float* pb = W + (size_t)kr * N + n0 + nb;

  v8f acc[8] = {};
  v4f ra0, ra1, ra2, ra3, rb0, rb1, rb2, rb3;

  auto fetch = [&](int k0) {
    const v4f* qa = reinterpret_cast<const v4f*>(pa + k0);
    ra0 = qa[0]; ra1 = qa[1]; ra2 = qa[2]; ra3 = qa[3];
    const v4f* qb = reinterpret_cast<const v4f*>(pb + (size_t)k0 * N);
    rb0 = qb[0]; rb1 = qb[1]; rb2 = qb[2]; rb3 = qb[3];
  };

  fetch(0);
  for (int k0 = 0; k0 < K; k0 += 32) {
    // registers -> LDS (with fp32->bf16 convert)
    *(v8bf*)&lA[rowA][c0]     = cvt8(ra0, ra1);
    *(v8bf*)&lA[rowA][c0 + 8] = cvt8(ra2, ra3);
    #pragma unroll
    for (int i = 0; i < 4; ++i) {
      lBt[nb + i][kr]      = (__bf16)rb0[i];
      lBt[nb + 4 + i][kr]  = (__bf16)rb1[i];
      lBt[nb + 8 + i][kr]  = (__bf16)rb2[i];
      lBt[nb + 12 + i][kr] = (__bf16)rb3[i];
    }
    __syncthreads();

    if (k0 + 32 < K) fetch(k0 + 32);            // overlap next slab with WMMAs

    v16bf a = combine16(*(const v8bf*)&lA[mrow][hsel],
                        *(const v8bf*)&lA[mrow][hsel + 16]);
    #pragma unroll
    for (int nt = 0; nt < 8; ++nt) {
      const int ncol = nt * 16 + col;
      v16bf b = combine16(*(const v8bf*)&lBt[ncol][hsel],
                          *(const v8bf*)&lBt[ncol][hsel + 16]);
      acc[nt] = __builtin_amdgcn_wmma_f32_16x16x32_bf16(
          false, a, false, b, (short)0, acc[nt], false, false);
    }
    __syncthreads();
  }

  #pragma unroll
  for (int nt = 0; nt < 8; ++nt) {
    const int n_g = n0 + nt * 16 + col;
    const float bv = bias[n_g];
    #pragma unroll
    for (int r = 0; r < 8; ++r) {
      const int m_g = m0 + 16 * wave + r + hsel;
      const float val = acc[nt][r] + bv;
      if (MODE == 0) {
        outF[(size_t)m_g * N + n_g] = val;
      } else {
        const int seg  = n_g >> 10;             // 0=q 1=k 2=v
        const int cn   = n_g & (Cc - 1);
        const int head = cn >> 6;
        const int d    = cn & (HDc - 1);
        const int b    = m_g >> 11;             // / T
        const int t    = m_g & (Tc - 1);
        const __bf16 bb = (__bf16)val;
        const size_t bh = (size_t)(b * NHc + head);
        if (seg == 0)      Qb[(bh * Tc + t) * HDc + d] = bb;
        else if (seg == 1) Kb[(bh * Tc + t) * HDc + d] = bb;
        else               Vt[(bh * HDc + d) * Tc + t] = bb;
      }
    }
  }
}

// ---------------------------------------------------------------------------
// Flash attention forward: per (b,h), causal, online softmax.
// 128 threads = 4 waves; each wave owns 16 query rows; key blocks of 64
// (softmax shuffles / accumulator rescale amortized over 2x keys).
// ---------------------------------------------------------------------------
__global__ __launch_bounds__(128)
void attn_fwd_kernel(const __bf16* __restrict__ Qb, const __bf16* __restrict__ Kb,
                     const __bf16* __restrict__ Vt, float* __restrict__ Y)
{
  __shared__ alignas(16) __bf16 lp[4][16][72];   // per-wave P buffer, 144B rows

  const int tid  = threadIdx.x;
  const int lane = tid & 31;
  const int wave = tid >> 5;
  const int bh = blockIdx.y;
  const int b  = bh >> 4;
  const int h  = bh & 15;
  const int q0 = blockIdx.x * 64 + wave * 16;
  const int col  = lane & 15;
  const int hsel = (lane >> 4) << 3;

  const __bf16* Qh = Qb + (size_t)bh * Tc * HDc;
  const __bf16* Kh = Kb + (size_t)bh * Tc * HDc;
  const __bf16* Vh = Vt + (size_t)bh * HDc * Tc;

  // Q fragments for this wave's 16 rows (reused across all key blocks)
  v16bf aq[2];
  #pragma unroll
  for (int c = 0; c < 2; ++c) {
    const __bf16* qp = Qh + (size_t)(q0 + col) * HDc + c * 32 + hsel;
    aq[c] = combine16(*(const v8bf*)qp, *(const v8bf*)(qp + 16));
  }

  v8f o[4] = {};
  float mstat[8], lstat[8];
  #pragma unroll
  for (int r = 0; r < 8; ++r) { mstat[r] = -3.0e38f; lstat[r] = 0.0f; }

  const float scale = 0.125f;                    // 1/sqrt(64)
  const int nkb = ((q0 + 15) >> 6) + 1;          // 64-key blocks, causal bound

  for (int kb = 0; kb < nkb; ++kb) {
    const int kbase = kb * 64;

    // S = Q @ K^T for four 16-key tiles
    v8f s[4] = {};
    #pragma unroll
    for (int sub = 0; sub < 4; ++sub) {
      const int key = kbase + sub * 16 + col;
      #pragma unroll
      for (int c = 0; c < 2; ++c) {
        const __bf16* kp = Kh + (size_t)key * HDc + c * 32 + hsel;
        v16bf bk = combine16(*(const v8bf*)kp, *(const v8bf*)(kp + 16));
        s[sub] = __builtin_amdgcn_wmma_f32_16x16x32_bf16(
            false, aq[c], false, bk, (short)0, s[sub], false, false);
      }
    }

    // scale + causal mask (D layout: lane col = key, rows r+hsel = query)
    #pragma unroll
    for (int sub = 0; sub < 4; ++sub) {
      #pragma unroll
      for (int r = 0; r < 8; ++r) {
        const float v  = s[sub][r] * scale;
        const int  kg  = kbase + sub * 16 + col;
        const int  qg  = q0 + r + hsel;
        s[sub][r] = (kg > qg) ? -3.0e38f : v;
      }
    }

    // online softmax: row reductions within 16-lane halves (p overwrites s)
    #pragma unroll
    for (int r = 0; r < 8; ++r) {
      float t = fmaxf(fmaxf(s[0][r], s[1][r]), fmaxf(s[2][r], s[3][r]));
      t = fmaxf(t, __shfl_xor(t, 1));
      t = fmaxf(t, __shfl_xor(t, 2));
      t = fmaxf(t, __shfl_xor(t, 4));
      t = fmaxf(t, __shfl_xor(t, 8));
      const float mn    = fmaxf(mstat[r], t);
      const float alpha = __expf(mstat[r] - mn);
      mstat[r] = mn;
      float rs = 0.0f;
      #pragma unroll
      for (int sub = 0; sub < 4; ++sub) {
        const float p = __expf(s[sub][r] - mn);
        s[sub][r] = p;
        rs += p;
      }
      rs += __shfl_xor(rs, 1);
      rs += __shfl_xor(rs, 2);
      rs += __shfl_xor(rs, 4);
      rs += __shfl_xor(rs, 8);
      lstat[r] = lstat[r] * alpha + rs;
      #pragma unroll
      for (int nt = 0; nt < 4; ++nt) o[nt][r] *= alpha;
    }

    // P: D-layout -> A-layout via per-wave private LDS (in-order DS ops)
    #pragma unroll
    for (int sub = 0; sub < 4; ++sub)
      #pragma unroll
      for (int r = 0; r < 8; ++r)
        lp[wave][r + hsel][sub * 16 + col] = (__bf16)s[sub][r];
    asm volatile("s_wait_dscnt 0x0" ::: "memory");

    v16bf ap[2];
    #pragma unroll
    for (int kc = 0; kc < 2; ++kc)
      ap[kc] = combine16(*(const v8bf*)&lp[wave][col][kc * 32 + hsel],
                         *(const v8bf*)&lp[wave][col][kc * 32 + hsel + 16]);

    // O += P @ V  (V pre-transposed [d][T]: contiguous keys per lane)
    #pragma unroll
    for (int nt = 0; nt < 4; ++nt) {
      #pragma unroll
      for (int kc = 0; kc < 2; ++kc) {
        const __bf16* vp = Vh + (size_t)(nt * 16 + col) * Tc + kbase + kc * 32 + hsel;
        v16bf bv = combine16(*(const v8bf*)vp, *(const v8bf*)(vp + 16));
        o[nt] = __builtin_amdgcn_wmma_f32_16x16x32_bf16(
            false, ap[kc], false, bv, (short)0, o[nt], false, false);
      }
    }
  }

  // normalize and write Y[b, t, h*64 + d] fp32
  float invl[8];
  #pragma unroll
  for (int r = 0; r < 8; ++r) invl[r] = 1.0f / lstat[r];
  #pragma unroll
  for (int nt = 0; nt < 4; ++nt) {
    #pragma unroll
    for (int r = 0; r < 8; ++r) {
      const int row = q0 + r + hsel;
      Y[((size_t)(b * Tc + row)) * Cc + h * HDc + nt * 16 + col] = o[nt][r] * invl[r];
    }
  }
}

extern "C" void kernel_launch(void* const* d_in, const int* in_sizes, int n_in,
                              void* d_out, int out_size, void* d_ws, size_t ws_size,
                              hipStream_t stream) {
  const float* x      = (const float*)d_in[0];
  const float* w_attn = (const float*)d_in[1];
  const float* b_attn = (const float*)d_in[2];
  const float* w_proj = (const float*)d_in[3];
  const float* b_proj = (const float*)d_in[4];
  float* out = (float*)d_out;

  const size_t HSZ = (size_t)Bc * NHc * Tc * HDc;   // 8,388,608 elems
  __bf16* Qb = (__bf16*)d_ws;
  __bf16* Kb = Qb + HSZ;
  __bf16* Vt = Kb + HSZ;
  float*  Yf = (float*)(Vt + HSZ);                  // 8192x1024 fp32

  // 1) QKV projection with fused bias + head-layout scatter (bf16)
  wmma_gemm_kernel<1><<<dim3(N3 / 128, Mrows / 128), 256, 0, stream>>>(
      x, w_attn, b_attn, nullptr, Qb, Kb, Vt, Mrows, N3, Cc);

  // 2) causal flash attention
  attn_fwd_kernel<<<dim3(Tc / 64, Bc * NHc), 128, 0, stream>>>(Qb, Kb, Vt, Yf);

  // 3) output projection + bias -> d_out (fp32)
  wmma_gemm_kernel<0><<<dim3(Cc / 128, Mrows / 128), 256, 0, stream>>>(
      Yf, w_proj, b_proj, out, nullptr, nullptr, nullptr, Mrows, Cc, Cc);
}
